// HGCN_36532991819877
// MI455X (gfx1250) — matrix operand order repaired and checked by reading, not compiled, not run on hardware
//
#include <hip/hip_runtime.h>
#include <math.h>

typedef __bf16 v16bf __attribute__((ext_vector_type(16)));
typedef float  v8f   __attribute__((ext_vector_type(8)));

#define DIM 128
#define LAYERS 2
#define RELS 2

__device__ __forceinline__ unsigned short f2bf(float f) {
  unsigned int u = __float_as_uint(f);
  // round-to-nearest-even f32 -> bf16
  return (unsigned short)((u + 0x7FFFu + ((u >> 16) & 1u)) >> 16);
}

union BVec {
  v16bf v;
  unsigned short s[16];
  uint4 u4[2];
};

// Transpose+convert: wt[mat][n*DIM+k] = bf16(W[mat][k*DIM+n]) for W1 and W2.
__global__ void prep_weights(const float* __restrict__ W1, const float* __restrict__ W2,
                             unsigned short* __restrict__ wt1, unsigned short* __restrict__ wt2,
                             int per /* LAYERS*RELS*DIM*DIM */) {
  int t = blockIdx.x * blockDim.x + threadIdx.x;
  if (t >= 2 * per) return;
  int which = (t >= per) ? 1 : 0;
  int rem = which ? (t - per) : t;
  int mat = rem / (DIM * DIM);
  int ij = rem - mat * (DIM * DIM);
  int n = ij >> 7;
  int k = ij & 127;
  const float* Wsrc = which ? W2 : W1;
  unsigned short* out = which ? wt2 : wt1;
  out[(size_t)mat * DIM * DIM + (size_t)n * DIM + k] =
      f2bf(Wsrc[(size_t)mat * DIM * DIM + (size_t)k * DIM + n]);
}

// h_cur = feat (f32) and h_bf = bf16(feat), fused.
__global__ void init_h(const float* __restrict__ feat, float* __restrict__ h_cur,
                       unsigned short* __restrict__ h_bf, int n) {
  int i = blockIdx.x * blockDim.x + threadIdx.x;
  if (i >= n) return;
  float v = feat[i];
  h_cur[i] = v;
  h_bf[i] = f2bf(v);
}

// z1 = h @ W1, z2 = h @ W2 via bf16 WMMA with f32 accumulation.
// One wave computes one 16x16 output tile for both weight matrices.
// A operand comes from the pre-converted bf16 mirror of h -> inner loop is
// pure b128 loads + v_wmma (no per-element conversion VALU).
__global__ void gemm_dual_bf16(const unsigned short* __restrict__ hb,  // [N][DIM] bf16 row-major
                               const unsigned short* __restrict__ wt1, // [DIM][DIM] transposed bf16
                               const unsigned short* __restrict__ wt2,
                               float* __restrict__ z1, float* __restrict__ z2, int N) {
  const int lane = threadIdx.x & 31;
  const int wave = threadIdx.x >> 5;   // 8 waves -> 8 column tiles (8*16 = 128)
  const int row0 = blockIdx.x * 16;
  const int col0 = wave * 16;
  const int r = lane & 15;
  const int hi = lane >> 4;
  const int rowA = row0 + r;
  const int colB = col0 + r;

  v8f acc1 = {};
  v8f acc2 = {};

  const unsigned short* arow = hb + (size_t)rowA * DIM;
  const unsigned short* brow1 = wt1 + (size_t)colB * DIM;
  const unsigned short* brow2 = wt2 + (size_t)colB * DIM;
  const bool arow_ok = (rowA < N);

#pragma unroll
  for (int k0 = 0; k0 < DIM; k0 += 32) {
    // A tile 16x32 bf16 (ISA 7.12.2): lane row = r;
    // elements 0..7  <- K = k0 + hi*8 + {0..7}
    // elements 8..15 <- K = k0 + 16 + hi*8 + {0..7}
    BVec a;
    if (arow_ok) {
      a.u4[0] = *(const uint4*)(arow + k0 + hi * 8);
      a.u4[1] = *(const uint4*)(arow + k0 + 16 + hi * 8);
    } else {
      a.u4[0] = make_uint4(0, 0, 0, 0);
      a.u4[1] = make_uint4(0, 0, 0, 0);
    }
    // B tile 32x16 bf16: lane col = r; same K order as A:
    // elements 0..7 <- K = k0 + hi*8 + {0..7}, elements 8..15 <- K = k0 + 16 + hi*8 + {0..7}
    BVec b1, b2;
    const unsigned short* bp1 = brow1 + k0 + hi * 8;
    const unsigned short* bp2 = brow2 + k0 + hi * 8;
    b1.u4[0] = *(const uint4*)(bp1);
    b1.u4[1] = *(const uint4*)(bp1 + 16);
    b2.u4[0] = *(const uint4*)(bp2);
    b2.u4[1] = *(const uint4*)(bp2 + 16);

    acc1 = __builtin_amdgcn_wmma_f32_16x16x32_bf16(false, a.v, false, b1.v, (short)0, acc1, false, false);
    acc2 = __builtin_amdgcn_wmma_f32_16x16x32_bf16(false, a.v, false, b2.v, (short)0, acc2, false, false);
  }

  // C/D layout: VGPR i -> M = i + 8*hi, N = lane&15
#pragma unroll
  for (int i = 0; i < 8; i++) {
    int rr = row0 + hi * 8 + i;
    if (rr < N) {
      z1[(size_t)rr * DIM + col0 + r] = acc1[i];
      z2[(size_t)rr * DIM + col0 + r] = acc2[i];
    }
  }
}

__global__ void fill_f(float* __restrict__ p, float v, int n) {
  int i = blockIdx.x * blockDim.x + threadIdx.x;
  if (i < n) p[i] = v;
}

// el[i] = z1[i,:] . a[:128]; er[i] = z2[i,:] . a[128:]
__global__ void attn_dots(const float* __restrict__ z1, const float* __restrict__ z2,
                          const float* __restrict__ a,
                          float* __restrict__ el, float* __restrict__ er, int N) {
  int i = blockIdx.x * blockDim.x + threadIdx.x;
  if (i >= N) return;
  const float* r1 = z1 + (size_t)i * DIM;
  const float* r2 = z2 + (size_t)i * DIM;
  float s1 = 0.f, s2 = 0.f;
#pragma unroll 8
  for (int j = 0; j < DIM; j++) {
    s1 += r1[j] * a[j];
    s2 += r2[j] * a[DIM + j];
  }
  el[i] = s1;
  er[i] = s2;
}

// float atomic max via signed-int max / unsigned-int min trick (sign-safe)
__device__ __forceinline__ void atomicMaxF(float* addr, float val) {
  if (val >= 0.0f)
    atomicMax((int*)addr, __float_as_int(val));
  else
    atomicMin((unsigned int*)addr, __float_as_uint(val));
}

__global__ void edge_max(const int* __restrict__ src, const int* __restrict__ dst,
                         const float* __restrict__ el, const float* __restrict__ er,
                         float* __restrict__ e, float* __restrict__ m, int E) {
  int k = blockIdx.x * blockDim.x + threadIdx.x;
  if (k >= E) return;
  float v = el[src[k]] + er[dst[k]];
  v = (v > 0.f) ? v : 0.01f * v;   // leaky_relu
  e[k] = v;
  atomicMaxF(&m[dst[k]], v);
}

__global__ void fix_m(float* __restrict__ m, int N) {
  int i = blockIdx.x * blockDim.x + threadIdx.x;
  if (i >= N) return;
  float v = m[i];
  m[i] = (v < -1e37f) ? 0.f : v;   // empty segments (-inf) -> 0
}

__global__ void edge_expsum(const int* __restrict__ dst, float* __restrict__ e,
                            const float* __restrict__ m, float* __restrict__ denom, int E) {
  int k = blockIdx.x * blockDim.x + threadIdx.x;
  if (k >= E) return;
  int d = dst[k];
  float ex = __expf(e[k] - m[d]);
  e[k] = ex;
  atomicAdd(&denom[d], ex);
}

// One wave per edge: accum[dst,:] += alpha * z_src[src,:]
__global__ void edge_scatter(const int* __restrict__ src, const int* __restrict__ dst,
                             const float* __restrict__ e, const float* __restrict__ denom,
                             const float* __restrict__ zs, float* __restrict__ accum, int E) {
  int widx = (blockIdx.x * blockDim.x + threadIdx.x) >> 5;
  int lane = threadIdx.x & 31;
  if (widx >= E) return;
  int d = dst[widx];
  int s = src[widx];
  float den = denom[d];
  float alpha = e[widx] / ((den == 0.f) ? 1.f : den);
  const float* zrow = zs + (size_t)s * DIM;
  float* orow = accum + (size_t)d * DIM;
#pragma unroll
  for (int j = lane; j < DIM; j += 32)
    atomicAdd(&orow[j], alpha * zrow[j]);
}

// h = relu(accum / RELS), fused with refresh of the bf16 mirror for next layer's GEMM
__global__ void finalize(const float* __restrict__ accum, float* __restrict__ h,
                         unsigned short* __restrict__ h_bf, int n) {
  int i = blockIdx.x * blockDim.x + threadIdx.x;
  if (i >= n) return;
  float v = accum[i] * (1.0f / (float)RELS);
  v = (v > 0.f) ? v : 0.f;
  h[i] = v;
  h_bf[i] = f2bf(v);
}

__global__ void classifier(const float* __restrict__ h,
                           const int* __restrict__ ps, const int* __restrict__ pd,
                           const float* __restrict__ Wc, const float* __restrict__ bc,
                           float* __restrict__ out, int P) {
  int p = blockIdx.x * blockDim.x + threadIdx.x;
  if (p >= P) return;
  const float* hs = h + (size_t)ps[p] * DIM;
  const float* hd = h + (size_t)pd[p] * DIM;
  float a0 = bc[0], a1 = bc[1];
#pragma unroll 8
  for (int j = 0; j < DIM; j++) {
    float v = hs[j];
    a0 += v * Wc[2 * j];
    a1 += v * Wc[2 * j + 1];
  }
#pragma unroll 8
  for (int j = 0; j < DIM; j++) {
    float v = hd[j];
    a0 += v * Wc[2 * (DIM + j)];
    a1 += v * Wc[2 * (DIM + j) + 1];
  }
  out[2 * p]     = 1.f / (1.f + __expf(-a0));
  out[2 * p + 1] = 1.f / (1.f + __expf(-a1));
}

extern "C" void kernel_launch(void* const* d_in, const int* in_sizes, int n_in,
                              void* d_out, int out_size, void* d_ws, size_t ws_size,
                              hipStream_t stream) {
  const float* feat = (const float*)d_in[0];
  const float* W1   = (const float*)d_in[1];
  const float* W2   = (const float*)d_in[2];
  const float* A    = (const float*)d_in[3];
  const float* Wc   = (const float*)d_in[4];
  const float* bc   = (const float*)d_in[5];
  const int* src_idx  = (const int*)d_in[6];
  const int* dst_idx  = (const int*)d_in[7];
  const int* pair_src = (const int*)d_in[8];
  const int* pair_dst = (const int*)d_in[9];

  const int N = in_sizes[0] / DIM;
  const int E = in_sizes[6] / RELS;
  const int P = in_sizes[8];
  const size_t ND = (size_t)N * DIM;

  // Workspace layout (floats)
  float* ws    = (float*)d_ws;
  float* h_cur = ws;
  float* z_src = h_cur + ND;
  float* z_dst = z_src + ND;
  float* accum = z_dst + ND;
  float* el    = accum + ND;
  float* er    = el + N;
  float* m     = er + N;
  float* denom = m + N;
  float* ebuf  = denom + N;
  unsigned short* wt1  = (unsigned short*)(ebuf + E);
  unsigned short* wt2  = wt1 + (size_t)LAYERS * RELS * DIM * DIM;
  unsigned short* h_bf = wt2 + (size_t)LAYERS * RELS * DIM * DIM;  // [N][DIM] bf16

  const int BS = 256;
  const int per = LAYERS * RELS * DIM * DIM;

  // Weight prep (bf16, transposed) + h init (f32 + bf16 mirror)
  prep_weights<<<(2 * per + BS - 1) / BS, BS, 0, stream>>>(W1, W2, wt1, wt2, per);
  init_h<<<((int)ND + BS - 1) / BS, BS, 0, stream>>>(feat, h_cur, h_bf, (int)ND);

  for (int l = 0; l < LAYERS; l++) {
    fill_f<<<((int)ND + BS - 1) / BS, BS, 0, stream>>>(accum, 0.f, (int)ND);
    for (int r = 0; r < RELS; r++) {
      const int mat = l * RELS + r;
      const unsigned short* w1p = wt1 + (size_t)mat * DIM * DIM;
      const unsigned short* w2p = wt2 + (size_t)mat * DIM * DIM;
      gemm_dual_bf16<<<(N + 15) / 16, BS, 0, stream>>>(h_bf, w1p, w2p, z_src, z_dst, N);
      attn_dots<<<(N + BS - 1) / BS, BS, 0, stream>>>(z_src, z_dst, A + (size_t)mat * 2 * DIM,
                                                      el, er, N);
      fill_f<<<(N + BS - 1) / BS, BS, 0, stream>>>(m, -INFINITY, N);
      fill_f<<<(N + BS - 1) / BS, BS, 0, stream>>>(denom, 0.f, N);
      const int* sidx = src_idx + (size_t)r * E;
      const int* didx = dst_idx + (size_t)r * E;
      edge_max<<<(E + BS - 1) / BS, BS, 0, stream>>>(sidx, didx, el, er, ebuf, m, E);
      fix_m<<<(N + BS - 1) / BS, BS, 0, stream>>>(m, N);
      edge_expsum<<<(E + BS - 1) / BS, BS, 0, stream>>>(didx, ebuf, m, denom, E);
      // one wave (32 lanes) per edge
      long long sthreads = (long long)E * 32;
      edge_scatter<<<(int)((sthreads + BS - 1) / BS), BS, 0, stream>>>(sidx, didx, ebuf, denom,
                                                                       z_src, accum, E);
    }
    finalize<<<((int)ND + BS - 1) / BS, BS, 0, stream>>>(accum, h_cur, h_bf, (int)ND);
  }

  // Output: [h (N*DIM floats)] then [sigmoid(logits) (P*2 floats)]
  float* out_h = (float*)d_out;
  float* out_p = out_h + ND;
  hipMemcpyAsync(out_h, h_cur, ND * sizeof(float), hipMemcpyDeviceToDevice, stream);
  classifier<<<(P + BS - 1) / BS, BS, 0, stream>>>(h_cur, pair_src, pair_dst, Wc, bc, out_p, P);
}
